// KStackModel_34497177321815
// MI455X (gfx1250) — matrix operand order repaired and checked by reading, not compiled, not run on hardware
//
#include <hip/hip_runtime.h>
#include <cstdint>
#include <cstddef>

// ---------------------------------------------------------------------------
// K2Layer forward for MI455X (gfx1250, wave32, WMMA + TDM).
//   B=8, W=1024, D=1024, R=32
// Pipeline:
//   1. convert weights / (gate*k_base) / u to bf16 in workspace
//   2. rmsnorm1 + xv = h_norm @ v          (fp32 VALU, row kernels)
//   3. gamma scan -> mixed (linear recurrence, replaces [R,W,W] decay tensor)
//   4. GEMM1: out  = (gate*kb) @ h_norm  + (alpha*mixed) @ u^T      [WMMA bf16]
//   5. GEMM2: h1   = h + out @ proj_w^T + proj_b  -> d_out          [WMMA bf16]
//   6. rmsnorm2(h1) -> h2 (bf16)
//   7. GEMM3: g    = gelu(h2 @ up_w^T + up_b)                       [WMMA bf16]
//   8. GEMM4: d_out = h1 + g @ down_w^T + down_b                    [WMMA bf16]
// Tile staging of row-contiguous operands goes through the Tensor Data Mover
// (tensor_load_to_lds + s_wait_tensorcnt) when the builtin is available.
// ---------------------------------------------------------------------------

#define B_ 8
#define W_ 1024
#define D_ 1024
#define R_ 32

typedef __attribute__((ext_vector_type(16))) __bf16 v16bf;
typedef __attribute__((ext_vector_type(8)))  __bf16 v8bf;
typedef __attribute__((ext_vector_type(8)))  float  v8f;
typedef __attribute__((ext_vector_type(4)))  unsigned int u32x4;
typedef __attribute__((ext_vector_type(8)))  int i32x8;
typedef __attribute__((ext_vector_type(4)))  int i32x4;

#if defined(__HIP_DEVICE_COMPILE__) && __has_builtin(__builtin_amdgcn_tensor_load_to_lds)
#define HAVE_TDM 1
#else
#define HAVE_TDM 0
#endif

__device__ __forceinline__ unsigned short f2bf(float f) {
  unsigned u = __float_as_uint(f);
  unsigned r = u + 0x7FFFu + ((u >> 16) & 1u);   // round-to-nearest-even
  return (unsigned short)(r >> 16);
}
__device__ __forceinline__ float sigmoidf(float x) { return 1.0f / (1.0f + expf(-x)); }

#if HAVE_TDM
// Issue one TDM 2D-tile load: 128 rows x 64B (32 bf16) from row-major global
// (row stride ld_elems bf16) into LDS with 16B padding after every 64B row
// (=> LDS row stride 80B = 40 ushorts, matching the fragment addressing).
// Descriptor per CDNA5 ISA ch.8: D# group0 (128b) + group1 (256b), 2D tile.
__device__ __forceinline__ void tdm_load_tile(const unsigned short* gsrc, int ld_elems,
                                              const unsigned short* lds_dst) {
  unsigned long long ga = (unsigned long long)(uintptr_t)gsrc;
  unsigned lds = (unsigned)(uintptr_t)lds_dst;       // LDS aperture: low 32 bits = LDS offset
  u32x4 g0;
  g0.x = 1u;                                         // count=1, user descriptor, no gather
  g0.y = lds;                                        // lds_addr
  g0.z = (unsigned)ga;                               // global_addr[31:0]
  g0.w = (unsigned)(ga >> 32) | 0x80000000u;         // global_addr[56:32] | type=2
  // data_size=3 (8B units) | pad_enable | pad_interval=3 (64B) | pad_amount=3 (16B)
  const unsigned td0 = 8, td1 = 128, tile0 = 8, tile1 = 128;
  unsigned long long s0 = (unsigned long long)(unsigned)(ld_elems >> 2);  // row stride, 8B units
  i32x8 g1;
  g1[0] = (int)((3u << 16) | (1u << 20) | (3u << 22) | (3u << 25));
  g1[1] = (int)((td0 & 0xFFFFu) << 16);              // tensor_dim0[15:0]
  g1[2] = (int)((td0 >> 16) | ((td1 & 0xFFFFu) << 16));
  g1[3] = (int)((td1 >> 16) | (tile0 << 16));        // tile_dim0
  g1[4] = (int)tile1;                                // tile_dim1 (tile_dim2=0)
  g1[5] = (int)(unsigned)(s0 & 0xFFFFFFFFu);         // tensor_dim0_stride[31:0]
  g1[6] = (int)(unsigned)(s0 >> 32);                 // stride[47:32] | dim1_stride[15:0]=0
  g1[7] = 0;
  i32x4 z4 = {0, 0, 0, 0};
#if defined(__clang_major__) && (__clang_major__ >= 23)
  i32x8 z8 = {0, 0, 0, 0, 0, 0, 0, 0};
  __builtin_amdgcn_tensor_load_to_lds(g0, g1, z4, z4, z8, 0);
#else
  __builtin_amdgcn_tensor_load_to_lds(g0, g1, z4, z4, 0);
#endif
}
#endif  // HAVE_TDM

// ------------------------------- converters --------------------------------

__global__ __launch_bounds__(256) void cvt_bf16(const float* __restrict__ src,
                                                unsigned short* __restrict__ dst, int n) {
  int i = blockIdx.x * 256 + threadIdx.x;
  if (i < n) dst[i] = f2bf(src[i]);
}

// dst[t,s] = (s<=t) ? sigmoid(gate_logit)*k_base[t,s] : 0   (gate folded in)
__global__ __launch_bounds__(256) void cvt_kbase(const float* __restrict__ src,
                                                 unsigned short* __restrict__ dst,
                                                 const float* __restrict__ gate_logit) {
  int i = blockIdx.x * 256 + threadIdx.x;
  if (i >= W_ * W_) return;
  int t = i >> 10, s = i & (W_ - 1);
  float g = sigmoidf(gate_logit[0]);
  dst[i] = (s <= t) ? f2bf(g * src[i]) : (unsigned short)0;
}

// ------------------------- rmsnorm (+ xv) kernels --------------------------

__global__ __launch_bounds__(256)
void rmsnorm_xv(const float* __restrict__ h, const float* __restrict__ scale,
                const float* __restrict__ v, unsigned short* __restrict__ hn_bf,
                float* __restrict__ xv) {
  __shared__ float sh[D_];
  __shared__ float red[256];
  const int row = blockIdx.x;            // b*W + t
  const int tid = threadIdx.x;
  const float* hr = h + (size_t)row * D_;

  float vals[4]; float ss = 0.f;
  #pragma unroll
  for (int i = 0; i < 4; ++i) { float x = hr[tid + i * 256]; vals[i] = x; ss += x * x; }
  red[tid] = ss; __syncthreads();
  for (int s = 128; s > 0; s >>= 1) { if (tid < s) red[tid] += red[tid + s]; __syncthreads(); }
  float inv = rsqrtf(red[0] / (float)D_ + 1e-8f);

  #pragma unroll
  for (int i = 0; i < 4; ++i) {
    int d = tid + i * 256;
    float hv = vals[i] * inv * scale[d];
    sh[d] = hv;
    hn_bf[(size_t)row * D_ + d] = f2bf(hv);
  }
  __syncthreads();

  // xv[row, r] = sum_d hn[d] * v[d, r]
  int r = tid & 31, chunk = tid >> 5;    // 8 chunks of 128
  float p = 0.f;
  for (int i = 0; i < 128; ++i) { int d = chunk * 128 + i; p += sh[d] * v[d * R_ + r]; }
  red[tid] = p; __syncthreads();
  if (tid < 32) {
    float s = 0.f;
    #pragma unroll
    for (int c = 0; c < 8; ++c) s += red[c * 32 + tid];
    xv[(size_t)row * R_ + tid] = s;
  }
}

__global__ __launch_bounds__(256)
void rmsnorm2(const float* __restrict__ h1, const float* __restrict__ scale,
              unsigned short* __restrict__ out_bf) {
  __shared__ float red[256];
  const int row = blockIdx.x;
  const int tid = threadIdx.x;
  const float* hr = h1 + (size_t)row * D_;
  float vals[4]; float ss = 0.f;
  #pragma unroll
  for (int i = 0; i < 4; ++i) { float x = hr[tid + i * 256]; vals[i] = x; ss += x * x; }
  red[tid] = ss; __syncthreads();
  for (int s = 128; s > 0; s >>= 1) { if (tid < s) red[tid] += red[tid + s]; __syncthreads(); }
  float inv = rsqrtf(red[0] / (float)D_ + 1e-8f);
  #pragma unroll
  for (int i = 0; i < 4; ++i) {
    int d = tid + i * 256;
    out_bf[(size_t)row * D_ + d] = f2bf(vals[i] * inv * scale[d]);
  }
}

// -------------------------- gamma-decay scan -------------------------------
// mixed[b,t,r] = gamma_r * mixed[b,t-1,r] + xv[b,t,r]; store alpha*mixed as bf16
__global__ __launch_bounds__(32)
void scan_mixed(const float* __restrict__ xv, const float* __restrict__ decay_logit,
                const float* __restrict__ alpha_logit, unsigned short* __restrict__ mixed_bf) {
  int b = blockIdx.x, r = threadIdx.x;
  float g = 0.15f + 0.85f * sigmoidf(decay_logit[r]);
  float alpha = sigmoidf(alpha_logit[0]);       // ALPHA_CAP = 1
  float acc = 0.f;
  const float* x = xv + (size_t)b * W_ * R_;
  unsigned short* m = mixed_bf + (size_t)b * W_ * R_;
  for (int t = 0; t < W_; ++t) {
    acc = g * acc + x[t * R_ + r];
    m[t * R_ + r] = f2bf(alpha * acc);
  }
}

// ------------------------------ WMMA GEMM ----------------------------------
// C[M,N] = A[M,K] (row-major bf16) @ B (bf16), tile 128x128, K-step 32,
// 8 wave32 waves (2 in M x 4 in N), each wave: 4x2 accumulators of 16x16.
// BMODE: BM_WEIGHT   -> B(k,n) = Wt[n*ldb + k]  (weight^T; TDM-stageable)
//        BM_ROWMAJOR -> B(k,n) = S [k*ldb + n]  (manual transposing stage)
// LDS B layout is always [n][k] so B fragments match the ISA lane striping.
enum { BM_WEIGHT = 0, BM_ROWMAJOR = 1 };
enum { EPI_BF16 = 0, EPI_H1 = 1, EPI_GELU = 2, EPI_FINAL = 3 };

#define LSTR 40   // padded LDS row stride (ushorts) == TDM pad 16B per 64B row

template <int BMODE, int EPI, bool EXTRA>
__global__ __launch_bounds__(256)
void gemm_wmma(const unsigned short* __restrict__ A, long strideAz, int lda,
               const unsigned short* __restrict__ Bm, long strideBz, int ldb,
               int M, int N, int K,
               const unsigned short* __restrict__ A2, long strideA2z,
               const unsigned short* __restrict__ B2,
               const float* __restrict__ addsrc,
               const float* __restrict__ bias,
               unsigned short* __restrict__ dst_bf, float* __restrict__ dst_f,
               long strideDz) {
  __shared__ unsigned short As[2][128 * LSTR];
  __shared__ unsigned short Bs[2][128 * LSTR];

  const int tid  = threadIdx.x;
  const int lane = tid & 31, wave = tid >> 5;
  const int wm = wave >> 2, wn = wave & 3;       // 2 x 4 wave grid
  const int z = blockIdx.z;
  const int mBase = blockIdx.y * 128, nBase = blockIdx.x * 128;
  const unsigned short* Ab = A + (size_t)z * strideAz;
  const unsigned short* Bb = Bm + (size_t)z * strideBz;

  // --- manual staging: 128 rows x 32 bf16 (64B/row), 2 x uint4 per thread ---
  auto stageA = [&](int buf, const unsigned short* src, int ld, int m0, int k0) {
    int row = tid >> 1, c = (tid & 1) * 16;
    const unsigned short* g = src + (size_t)(m0 + row) * ld + k0 + c;
    __builtin_prefetch(g + 64 * (size_t)ld, 0, 1);   // -> global_prefetch_b8
    const uint4* gp = reinterpret_cast<const uint4*>(g);
    uint4 x0 = gp[0], x1 = gp[1];
    uint4* s = reinterpret_cast<uint4*>(&As[buf][row * LSTR + c]);
    s[0] = x0; s[1] = x1;
  };
  auto stageBw = [&](int buf, const unsigned short* src, int ld, int n0, int k0) {
    int row = tid >> 1, c = (tid & 1) * 16;
    const uint4* gp = reinterpret_cast<const uint4*>(src + (size_t)(n0 + row) * ld + k0 + c);
    uint4 x0 = gp[0], x1 = gp[1];
    uint4* s = reinterpret_cast<uint4*>(&Bs[buf][row * LSTR + c]);
    s[0] = x0; s[1] = x1;
  };
  auto stageBrm = [&](int buf, const unsigned short* src, int ld, int k0, int n0) {
    int k = tid >> 3;                 // 0..31
    int c = (tid & 7) * 16;           // 0..112
    const uint4* gp = reinterpret_cast<const uint4*>(src + (size_t)(k0 + k) * ld + n0 + c);
    uint4 x0 = gp[0], x1 = gp[1];
    unsigned short vv[16];
    *reinterpret_cast<uint4*>(vv)     = x0;
    *reinterpret_cast<uint4*>(vv + 8) = x1;
    #pragma unroll
    for (int i = 0; i < 16; ++i) Bs[buf][(c + i) * LSTR + k] = vv[i];  // transpose to [n][k]
  };

  v8f acc[4][2];
  #pragma unroll
  for (int mi = 0; mi < 4; ++mi)
    #pragma unroll
    for (int ni = 0; ni < 2; ++ni)
      #pragma unroll
      for (int j = 0; j < 8; ++j) acc[mi][ni][j] = 0.f;

  // --- fragment load + 8 WMMAs from buffer `buf` ---
  auto compute = [&](int buf) {
    v16bf afr[4], bfr[2];
    #pragma unroll
    for (int mi = 0; mi < 4; ++mi) {
      int r  = wm * 64 + mi * 16 + (lane & 15);
      int kk = (lane >> 4) * 8;                      // A: elems0-7 -> K=kk..+7, elems8-15 -> +16
      v8bf lo = *reinterpret_cast<const v8bf*>(&As[buf][r * LSTR + kk]);
      v8bf hi = *reinterpret_cast<const v8bf*>(&As[buf][r * LSTR + kk + 16]);
      afr[mi] = __builtin_shufflevector(lo, hi, 0,1,2,3,4,5,6,7,8,9,10,11,12,13,14,15);
    }
    #pragma unroll
    for (int ni = 0; ni < 2; ++ni) {
      int n  = wn * 32 + ni * 16 + (lane & 15);
      int kk = (lane >> 4) * 8;                      // B: elems0-7 -> K=kk..+7, elems8-15 -> +16
      v8bf lo = *reinterpret_cast<const v8bf*>(&Bs[buf][n * LSTR + kk]);
      v8bf hi = *reinterpret_cast<const v8bf*>(&Bs[buf][n * LSTR + kk + 16]);
      bfr[ni] = __builtin_shufflevector(lo, hi, 0,1,2,3,4,5,6,7,8,9,10,11,12,13,14,15);
    }
    #pragma unroll
    for (int mi = 0; mi < 4; ++mi)
      #pragma unroll
      for (int ni = 0; ni < 2; ++ni)
        acc[mi][ni] = __builtin_amdgcn_wmma_f32_16x16x32_bf16(
            false, afr[mi], false, bfr[ni], (short)0, acc[mi][ni], false, false);
  };

  // --- main K loop, double-buffered; TDM stages row-contiguous tiles ---
  const int ksteps = K >> 5;
#if HAVE_TDM
  if (wave == 0) {
    tdm_load_tile(Ab + (size_t)mBase * lda, lda, &As[0][0]);
    if (BMODE == BM_WEIGHT) tdm_load_tile(Bb + (size_t)nBase * ldb, ldb, &Bs[0][0]);
  }
#else
  stageA(0, Ab, lda, mBase, 0);
  if (BMODE == BM_WEIGHT) stageBw(0, Bb, ldb, nBase, 0);
#endif
  if (BMODE == BM_ROWMAJOR) stageBrm(0, Bb, ldb, 0, nBase);

  int buf = 0;
  for (int kt = 0; kt < ksteps; ++kt) {
#if HAVE_TDM
    if (wave == 0) __builtin_amdgcn_s_wait_tensorcnt(0);  // current buf's TDM done
#endif
    __syncthreads();
    if (kt + 1 < ksteps) {
#if HAVE_TDM
      if (wave == 0) {   // DMA next tiles; overlaps with this tile's WMMAs
        tdm_load_tile(Ab + (size_t)mBase * lda + (size_t)(kt + 1) * 32, lda,
                      &As[buf ^ 1][0]);
        if (BMODE == BM_WEIGHT)
          tdm_load_tile(Bb + (size_t)nBase * ldb + (size_t)(kt + 1) * 32, ldb,
                        &Bs[buf ^ 1][0]);
      }
#else
      stageA(buf ^ 1, Ab, lda, mBase, (kt + 1) * 32);
      if (BMODE == BM_WEIGHT) stageBw(buf ^ 1, Bb, ldb, nBase, (kt + 1) * 32);
#endif
      if (BMODE == BM_ROWMAJOR) stageBrm(buf ^ 1, Bb, ldb, (kt + 1) * 32, nBase);
    } else if (EXTRA) {
      // extra rank-32 pass: A2 = alpha*mixed [M x 32] row-major, B2 = u^T (weight mode)
      stageA (buf ^ 1, A2 + (size_t)z * strideA2z, 32, mBase, 0);
      stageBw(buf ^ 1, B2, 32, nBase, 0);
    }
    compute(buf);
    buf ^= 1;
  }
  if (EXTRA) { __syncthreads(); compute(buf); }

  // --- epilogue ---
  #pragma unroll
  for (int mi = 0; mi < 4; ++mi)
    #pragma unroll
    for (int ni = 0; ni < 2; ++ni)
      #pragma unroll
      for (int j = 0; j < 8; ++j) {
        int row = mBase + wm * 64 + mi * 16 + (lane >> 4) * 8 + j;
        int col = nBase + wn * 32 + ni * 16 + (lane & 15);
        size_t gi = (size_t)z * strideDz + (size_t)row * N + col;
        float val = acc[mi][ni][j];
        if constexpr (EPI == EPI_BF16) {
          dst_bf[gi] = f2bf(val);
        } else if constexpr (EPI == EPI_H1) {
          dst_f[gi] = val + addsrc[gi] + bias[col];
        } else if constexpr (EPI == EPI_GELU) {
          float x = val + bias[col];
          float t = tanhf(0.7978845608028654f * (x + 0.044715f * x * x * x));
          dst_bf[gi] = f2bf(0.5f * x * (1.f + t));
        } else { // EPI_FINAL
          dst_f[gi] = dst_f[gi] + val + bias[col];
        }
      }
}

// ------------------------------- launcher ----------------------------------

extern "C" void kernel_launch(void* const* d_in, const int* in_sizes, int n_in,
                              void* d_out, int out_size, void* d_ws, size_t ws_size,
                              hipStream_t stream) {
  (void)in_sizes; (void)n_in; (void)out_size; (void)ws_size;
  const float* h           = (const float*)d_in[0];
  const float* k_base      = (const float*)d_in[1];
  const float* decay_logit = (const float*)d_in[2];
  const float* u           = (const float*)d_in[3];
  const float* v           = (const float*)d_in[4];
  const float* alpha_logit = (const float*)d_in[5];
  const float* gate_logit  = (const float*)d_in[6];
  const float* proj_w      = (const float*)d_in[7];
  const float* proj_b      = (const float*)d_in[8];
  const float* norm1_scale = (const float*)d_in[9];
  const float* norm2_scale = (const float*)d_in[10];
  const float* up_w        = (const float*)d_in[11];
  const float* up_b        = (const float*)d_in[12];
  const float* down_w      = (const float*)d_in[13];
  const float* down_b      = (const float*)d_in[14];
  float* out = (float*)d_out;

  const size_t BW  = (size_t)B_ * W_;
  const size_t BWD = BW * D_;

  char* ws = (char*)d_ws;
  size_t off = 0;
  auto alloc = [&](size_t bytes) -> void* {
    void* p = ws + off; off = (off + bytes + 255) & ~(size_t)255; return p;
  };
  unsigned short* hn_bf = (unsigned short*)alloc(BWD * 2);        // reused as h2 after GEMM2
  unsigned short* out_bf = (unsigned short*)alloc(BWD * 2);
  unsigned short* g_bf   = (unsigned short*)alloc(BWD * 4);       // [BW, 2D]
  unsigned short* kb_bf  = (unsigned short*)alloc((size_t)W_ * W_ * 2);
  unsigned short* pw_bf  = (unsigned short*)alloc((size_t)D_ * D_ * 2);
  unsigned short* uw_bf  = (unsigned short*)alloc((size_t)2 * D_ * D_ * 2);
  unsigned short* dw_bf  = (unsigned short*)alloc((size_t)2 * D_ * D_ * 2);
  unsigned short* u_bf   = (unsigned short*)alloc((size_t)D_ * R_ * 2);
  float*          xv     = (float*)alloc(BW * R_ * 4);
  unsigned short* mx_bf  = (unsigned short*)alloc(BW * R_ * 2);

  // 1. bf16 conversions (gate folded into k_base)
  cvt_kbase<<<(W_ * W_) / 256, 256, 0, stream>>>(k_base, kb_bf, gate_logit);
  cvt_bf16<<<(D_ * D_) / 256, 256, 0, stream>>>(proj_w, pw_bf, D_ * D_);
  cvt_bf16<<<(2 * D_ * D_) / 256, 256, 0, stream>>>(up_w, uw_bf, 2 * D_ * D_);
  cvt_bf16<<<(2 * D_ * D_) / 256, 256, 0, stream>>>(down_w, dw_bf, 2 * D_ * D_);
  cvt_bf16<<<(D_ * R_) / 256, 256, 0, stream>>>(u, u_bf, D_ * R_);

  // 2. rmsnorm1 + xv
  rmsnorm_xv<<<(int)BW, 256, 0, stream>>>(h, norm1_scale, v, hn_bf, xv);

  // 3. gamma scan -> alpha*mixed (bf16)
  scan_mixed<<<B_, 32, 0, stream>>>(xv, decay_logit, alpha_logit, mx_bf);

  // 4. GEMM1 (per batch): out = kb@h_norm + mixed@u^T   -> out_bf
  gemm_wmma<BM_ROWMAJOR, EPI_BF16, true>
      <<<dim3(D_ / 128, W_ / 128, B_), 256, 0, stream>>>(
      kb_bf, 0L, W_, hn_bf, (long)(W_ * D_), D_, W_, D_, W_,
      mx_bf, (long)(W_ * R_), u_bf, nullptr, nullptr,
      out_bf, nullptr, (long)(W_ * D_));

  // 5. GEMM2: h1 = h + out@proj_w^T + proj_b  -> d_out (fp32)
  gemm_wmma<BM_WEIGHT, EPI_H1, false>
      <<<dim3(D_ / 128, (int)BW / 128, 1), 256, 0, stream>>>(
      out_bf, 0L, D_, pw_bf, 0L, D_, (int)BW, D_, D_,
      nullptr, 0L, nullptr, h, proj_b, nullptr, out, 0L);

  // 6. rmsnorm2(h1) -> h2 (bf16, reuses hn_bf)
  rmsnorm2<<<(int)BW, 256, 0, stream>>>(out, norm2_scale, hn_bf);

  // 7. GEMM3: g = gelu(h2@up_w^T + up_b) -> g_bf
  gemm_wmma<BM_WEIGHT, EPI_GELU, false>
      <<<dim3(2 * D_ / 128, (int)BW / 128, 1), 256, 0, stream>>>(
      hn_bf, 0L, D_, uw_bf, 0L, D_, (int)BW, 2 * D_, D_,
      nullptr, 0L, nullptr, nullptr, up_b, g_bf, nullptr, 0L);

  // 8. GEMM4: d_out = h1 + g@down_w^T + down_b
  gemm_wmma<BM_WEIGHT, EPI_FINAL, false>
      <<<dim3(D_ / 128, (int)BW / 128, 1), 256, 0, stream>>>(
      g_bf, 0L, 2 * D_, dw_bf, 0L, 2 * D_, (int)BW, D_, 2 * D_,
      nullptr, 0L, nullptr, nullptr, down_b, nullptr, out, 0L);
}
